// AttHead_65901978189900
// MI455X (gfx1250) — matrix operand order repaired and checked by reading, not compiled, not run on hardware
//
#include <hip/hip_runtime.h>

typedef __bf16 bf16_t;
typedef __attribute__((ext_vector_type(16))) __bf16 v16bf;
typedef __attribute__((ext_vector_type(8)))  __bf16 v8bf;
typedef __attribute__((ext_vector_type(8)))  float  v8f;

#define B_ 8
#define T_ 2048
#define C_ 1024

// ---------- bf16 <-> f32 helpers (RNE, software; only used outside GEMM loops) ----------
static __device__ __forceinline__ bf16_t f2bf(float f) {
  unsigned u = __builtin_bit_cast(unsigned, f);
  unsigned r = u + 0x7FFFu + ((u >> 16) & 1u);
  unsigned short h = (unsigned short)(r >> 16);
  return __builtin_bit_cast(bf16_t, h);
}
static __device__ __forceinline__ float bf2f(bf16_t b) {
  unsigned short s = __builtin_bit_cast(unsigned short, b);
  unsigned u = ((unsigned)s) << 16;
  return __builtin_bit_cast(float, u);
}

// ---------- WMMA operand loaders (pure bf16, no conversion) ----------
// A operand (16x32 bf16): lane L -> row L%16; K chunks {0..7,16..23} (L<16),
// {8..15,24..31} (L>=16). Source row-major, leading dim ld (elements).
static __device__ __forceinline__ v16bf load_a_bf16(const bf16_t* src, int ld, int lane) {
  int r = lane & 15, half = lane >> 4;
  const bf16_t* p = src + (size_t)r * ld + half * 8;
  v8bf c0 = *(const v8bf*)(p);
  v8bf c1 = *(const v8bf*)(p + 16);
  v16bf out;
#pragma unroll
  for (int i = 0; i < 8; ++i) { out[i] = c0[i]; out[8 + i] = c1[i]; }
  return out;
}
// B operand (32x16 bf16) loaded from a row-major source whose ROW index is the
// output column (i.e. source holds B^T): lane n -> column n%16, K chunks
// {0..7,16..23} for lanes<16 and {8..15,24..31} for lanes>=16.
static __device__ __forceinline__ v16bf load_b_bf16(const bf16_t* src, int ld, int lane) {
  int r = lane & 15, half = lane >> 4;
  const bf16_t* p = src + (size_t)r * ld + half * 8;
  v8bf c0 = *(const v8bf*)(p);
  v8bf c1 = *(const v8bf*)(p + 16);
  v16bf out;
#pragma unroll
  for (int i = 0; i < 8; ++i) { out[i] = c0[i]; out[8 + i] = c1[i]; }
  return out;
}

static __device__ __forceinline__ v8f wmma_bf16(v16bf a, v16bf b, v8f c) {
  return __builtin_amdgcn_wmma_f32_16x16x32_bf16(false, a, false, b, (short)0, c,
                                                 false, false);
}

// ---------- Kernel 0: one-shot fp32 -> bf16 cast of x and the three W ----------
#define XN ((size_t)B_ * T_ * C_)    // 16,777,216
#define WN ((size_t)C_ * C_)         // 1,048,576 per matrix
__global__ __launch_bounds__(256) void convert_kernel(
    const float* __restrict__ x, const float* __restrict__ Wq,
    const float* __restrict__ Wk, const float* __restrict__ Wv,
    bf16_t* __restrict__ xb, bf16_t* __restrict__ Wb) {
  size_t off = ((size_t)blockIdx.x * 256 + threadIdx.x) * 8;
  const float* src;
  bf16_t* dst;
  if (off < XN) {
    src = x + off;
    dst = xb + off;
  } else {
    size_t o2 = off - XN;
    int sel = (int)(o2 / WN);
    size_t r = o2 - (size_t)sel * WN;
    src = ((sel == 0) ? Wq : (sel == 1) ? Wk : Wv) + r;
    dst = Wb + o2;
  }
  v8f v = *(const v8f*)src;
  v8bf o;
#pragma unroll
  for (int i = 0; i < 8; ++i) o[i] = f2bf(v[i]);
  *(v8bf*)dst = o;
}

// ---------- Kernel 1: projections. Y = xb @ W^T (all-bf16 operands) ----------
// sel 0 -> Qb (row major), 1 -> Kb (row major), 2 -> Vt (transposed [B][C][T]).
// Inner loop software-pipelined by one k-step: prefetch (k+32) operands before
// issuing this step's WMMAs, so loads overlap matrix math within the wave.
__global__ __launch_bounds__(256) void proj_kernel(
    const bf16_t* __restrict__ xb, const bf16_t* __restrict__ Wb,
    bf16_t* __restrict__ Qb, bf16_t* __restrict__ Kb, bf16_t* __restrict__ Vt) {
  int lane = threadIdx.x & 31;
  int wv = threadIdx.x >> 5;
  int sel = blockIdx.y;
  const bf16_t* W = Wb + (size_t)sel * WN;

  int flat = blockIdx.x * 8 + wv;   // 16384 waves: 1024 row-tiles x 16 col-groups
  int rowTile = flat >> 4;
  int cg = flat & 15;
  int m0 = rowTile * 16;            // row in flattened [B*T, C]
  int n0 = cg * 64;                 // 64 output channels per wave (4 wmma tiles)

  const bf16_t* arow = xb + (size_t)m0 * C_;
  const bf16_t* wrow = W + (size_t)n0 * C_;
  v8f acc[4] = {};

  v16bf a  = load_a_bf16(arow, C_, lane);
  v16bf b0 = load_b_bf16(wrow + (size_t)0 * 16 * C_, C_, lane);
  v16bf b1 = load_b_bf16(wrow + (size_t)1 * 16 * C_, C_, lane);
  v16bf b2 = load_b_bf16(wrow + (size_t)2 * 16 * C_, C_, lane);
  v16bf b3 = load_b_bf16(wrow + (size_t)3 * 16 * C_, C_, lane);

#pragma unroll 2
  for (int k0 = 0; k0 < C_; k0 += 32) {
    int kn = (k0 + 32) & (C_ - 1);  // last iteration: dummy wrap-around prefetch
    v16bf an  = load_a_bf16(arow + kn, C_, lane);
    v16bf bn0 = load_b_bf16(wrow + (size_t)0 * 16 * C_ + kn, C_, lane);
    v16bf bn1 = load_b_bf16(wrow + (size_t)1 * 16 * C_ + kn, C_, lane);
    v16bf bn2 = load_b_bf16(wrow + (size_t)2 * 16 * C_ + kn, C_, lane);
    v16bf bn3 = load_b_bf16(wrow + (size_t)3 * 16 * C_ + kn, C_, lane);
    acc[0] = wmma_bf16(a, b0, acc[0]);
    acc[1] = wmma_bf16(a, b1, acc[1]);
    acc[2] = wmma_bf16(a, b2, acc[2]);
    acc[3] = wmma_bf16(a, b3, acc[3]);
    a = an; b0 = bn0; b1 = bn1; b2 = bn2; b3 = bn3;
  }

  int r = lane & 15, half = lane >> 4;
  if (sel < 2) {
    bf16_t* Y = (sel == 0) ? Qb : Kb;
#pragma unroll
    for (int j = 0; j < 4; ++j)
#pragma unroll
      for (int i = 0; i < 8; ++i) {
        int m = half * 8 + i;
        Y[(size_t)(m0 + m) * C_ + n0 + 16 * j + r] = f2bf(acc[j][i]);
      }
  } else {
    int b_idx = m0 / T_;
    int t0 = m0 % T_;
#pragma unroll
    for (int j = 0; j < 4; ++j) {
      v8bf pack;
#pragma unroll
      for (int i = 0; i < 8; ++i) pack[i] = f2bf(acc[j][i]);
      bf16_t* dst = Vt + ((size_t)b_idx * C_ + n0 + 16 * j + r) * T_ + t0 + half * 8;
      *(v8bf*)dst = pack;
    }
  }
}

// ---------- Kernel 2: S = Q K^T * scale, lower-triangular tiles only ----------
__global__ __launch_bounds__(256) void scores_kernel(const bf16_t* __restrict__ Qb,
                                                     const bf16_t* __restrict__ Kb,
                                                     bf16_t* __restrict__ P) {
  int lane = threadIdx.x & 31;
  int wv = threadIdx.x >> 5;
  int batch = blockIdx.y;
  int flat = blockIdx.x * 8 + wv;   // 4096 waves: 128 q-tiles x 32 key-groups
  int qtile = flat >> 5;
  int kg = flat & 31;
  int q0 = qtile * 16;
  int k0 = kg * 64;
  if (k0 > q0) return;              // whole 16x64 strip above the diagonal

  // number of unmasked 16-key subtiles in this strip (wave-uniform)
  int jmax = ((q0 - k0) >> 4) + 1;
  if (jmax > 4) jmax = 4;

  const bf16_t* Qrow = Qb + ((size_t)batch * T_ + q0) * C_;
  const bf16_t* Krow = Kb + ((size_t)batch * T_ + k0) * C_;
  v8f acc[4] = {};

  v16bf a = load_a_bf16(Qrow, C_, lane);
  v16bf b[4];
#pragma unroll
  for (int j = 0; j < 4; ++j)
    if (j < jmax) b[j] = load_b_bf16(Krow + (size_t)(16 * j) * C_, C_, lane);

#pragma unroll 2
  for (int c0 = 0; c0 < C_; c0 += 32) {
    int cn = (c0 + 32) & (C_ - 1);  // last iteration: dummy wrap-around prefetch
    v16bf an = load_a_bf16(Qrow + cn, C_, lane);
    v16bf bn[4];
#pragma unroll
    for (int j = 0; j < 4; ++j)
      if (j < jmax) bn[j] = load_b_bf16(Krow + (size_t)(16 * j) * C_ + cn, C_, lane);
#pragma unroll
    for (int j = 0; j < 4; ++j)
      if (j < jmax) acc[j] = wmma_bf16(a, b[j], acc[j]);
    a = an;
#pragma unroll
    for (int j = 0; j < 4; ++j)
      if (j < jmax) b[j] = bn[j];
  }

  const float scale = 0.03125f;     // C^-0.5 = 1/sqrt(1024)
  int r = lane & 15, half = lane >> 4;
#pragma unroll
  for (int j = 0; j < 4; ++j)
#pragma unroll
    for (int i = 0; i < 8; ++i) {
      int m = half * 8 + i;
      P[((size_t)batch * T_ + q0 + m) * T_ + k0 + 16 * j + r] = f2bf(acc[j][i] * scale);
    }
}

// ---------- Kernel 3: causal softmax over each row of P (in place) ----------
__global__ __launch_bounds__(256) void softmax_kernel(bf16_t* __restrict__ P) {
  int row = blockIdx.x;             // batch*T + q
  int q = row & (T_ - 1);
  bf16_t* Prow = P + (size_t)row * T_;
  int tid = threadIdx.x;

  float m = -3.0e38f, s = 0.f;
  for (int k = tid; k <= q; k += 256) {
    float xv = bf2f(Prow[k]);
    float nm = fmaxf(m, xv);
    s = s * __expf(m - nm) + __expf(xv - nm);
    m = nm;
  }
  __shared__ float sm[256], ss[256];
  sm[tid] = m; ss[tid] = s;
  __syncthreads();
  for (int off = 128; off > 0; off >>= 1) {
    if (tid < off) {
      float m2 = sm[tid + off], s2 = ss[tid + off];
      float nm = fmaxf(sm[tid], m2);
      ss[tid] = ss[tid] * __expf(sm[tid] - nm) + s2 * __expf(m2 - nm);
      sm[tid] = nm;
    }
    __syncthreads();
  }
  float M = sm[0];
  float invS = 1.0f / ss[0];
  for (int k = tid; k < T_; k += 256) {
    float o = (k <= q) ? __expf(bf2f(Prow[k]) - M) * invS : 0.f;
    Prow[k] = f2bf(o);
  }
}

// ---------- Kernel 4: O = P V  (V stored transposed), causal-clipped K loop ----------
__global__ __launch_bounds__(256) void out_kernel(const bf16_t* __restrict__ P,
                                                  const bf16_t* __restrict__ Vt,
                                                  float* __restrict__ Out) {
  int lane = threadIdx.x & 31;
  int wv = threadIdx.x >> 5;
  int flat = blockIdx.x * 8 + wv;   // 16384 waves: 1024 row-tiles x 16 ch-groups
  int rowTile = flat >> 4;
  int cg = flat & 15;
  int m0 = rowTile * 16;
  int batch = m0 / T_;
  int q0 = m0 % T_;
  int c0 = cg * 64;

  const bf16_t* Prow = P + ((size_t)batch * T_ + q0) * T_;
  const bf16_t* Vrow = Vt + ((size_t)batch * C_ + c0) * T_;
  int kend = (q0 + 16 + 31) & ~31;  // causal bound, rounded to WMMA K step
  if (kend > T_) kend = T_;

  v8f acc[4] = {};

  v16bf a  = load_a_bf16(Prow, T_, lane);
  v16bf b0 = load_b_bf16(Vrow + (size_t)0 * 16 * T_, T_, lane);
  v16bf b1 = load_b_bf16(Vrow + (size_t)1 * 16 * T_, T_, lane);
  v16bf b2 = load_b_bf16(Vrow + (size_t)2 * 16 * T_, T_, lane);
  v16bf b3 = load_b_bf16(Vrow + (size_t)3 * 16 * T_, T_, lane);

#pragma unroll 2
  for (int k = 0; k < kend; k += 32) {
    int kn = k + 32;
    if (kn >= kend) kn = 0;         // last iteration: dummy wrap-around prefetch
    v16bf an  = load_a_bf16(Prow + kn, T_, lane);
    v16bf bn0 = load_b_bf16(Vrow + (size_t)0 * 16 * T_ + kn, T_, lane);
    v16bf bn1 = load_b_bf16(Vrow + (size_t)1 * 16 * T_ + kn, T_, lane);
    v16bf bn2 = load_b_bf16(Vrow + (size_t)2 * 16 * T_ + kn, T_, lane);
    v16bf bn3 = load_b_bf16(Vrow + (size_t)3 * 16 * T_ + kn, T_, lane);
    acc[0] = wmma_bf16(a, b0, acc[0]);
    acc[1] = wmma_bf16(a, b1, acc[1]);
    acc[2] = wmma_bf16(a, b2, acc[2]);
    acc[3] = wmma_bf16(a, b3, acc[3]);
    a = an; b0 = bn0; b1 = bn1; b2 = bn2; b3 = bn3;
  }

  int r = lane & 15, half = lane >> 4;
#pragma unroll
  for (int j = 0; j < 4; ++j)
#pragma unroll
    for (int i = 0; i < 8; ++i) {
      int m = half * 8 + i;
      Out[((size_t)batch * T_ + q0 + m) * C_ + c0 + 16 * j + r] = acc[j][i];
    }
}

extern "C" void kernel_launch(void* const* d_in, const int* in_sizes, int n_in,
                              void* d_out, int out_size, void* d_ws, size_t ws_size,
                              hipStream_t stream) {
  (void)in_sizes; (void)n_in; (void)out_size; (void)ws_size;
  const float* x  = (const float*)d_in[0];
  const float* Wk = (const float*)d_in[1];
  const float* Wq = (const float*)d_in[2];
  const float* Wv = (const float*)d_in[3];
  float* Out = (float*)d_out;

  // Workspace layout (160 MB total):
  //   [0   ..  32M) Qb   bf16 [B*T, C]
  //   [32M ..  64M) Kb   bf16 [B*T, C]
  //   [64M ..  96M) Vt   bf16 [B][C][T]
  //   [96M .. 160M) P    bf16 [B][T][T]   -- ALIASES xb/Wb (dead after proj)
  //   xb = 96M (32 MB), Wb = 128M (6 MB): written by convert, read by proj only.
  bf16_t* Qb = (bf16_t*)d_ws;
  bf16_t* Kb = Qb + XN;
  bf16_t* Vt = Kb + XN;
  bf16_t* P  = Vt + XN;
  bf16_t* xb = P;                   // alias: consumed before P is produced
  bf16_t* Wb = xb + XN;

  // convert: (XN + 3*WN)/8 = 2,490,368 threads -> 9728 blocks of 256
  convert_kernel<<<dim3(9728), 256, 0, stream>>>(x, Wq, Wk, Wv, xb, Wb);
  proj_kernel<<<dim3(2048, 3), 256, 0, stream>>>(xb, Wb, Qb, Kb, Vt);
  scores_kernel<<<dim3(512, 8), 256, 0, stream>>>(Qb, Kb, P);
  softmax_kernel<<<dim3(B_ * T_), 256, 0, stream>>>(P);
  out_kernel<<<dim3(2048), 256, 0, stream>>>(P, Vt, Out);
}
